// GeneralizedRCNNExtractModel_26036091748769
// MI455X (gfx1250) — matrix operand order repaired and checked by reading, not compiled, not run on hardware
//
#include <hip/hip_runtime.h>
#include <stdint.h>

// ---------------------------------------------------------------------------
// CDNA5 / gfx1250, wave32. BF16 WMMA GEMM pipeline with pre-swizzled operands
// and async global->LDS tile copies (ASYNCcnt) + double-buffered LDS.
// ---------------------------------------------------------------------------

typedef __attribute__((ext_vector_type(16))) __bf16 v16bf;
typedef __attribute__((ext_vector_type(8)))  float  v8f;

#define TILE_M 128
#define TILE_N 128
#define TILE_K 32
#define LDS_STRIDE 20

// pack two fp32 -> dword {lo=bf16(a), hi=bf16(b)}, round-to-nearest-even
static __device__ __forceinline__ uint32_t pk_bf16(float a, float b) {
    uint32_t ua = __float_as_uint(a);
    uint32_t ub = __float_as_uint(b);
    ua += 0x7FFFu + ((ua >> 16) & 1u);
    ub += 0x7FFFu + ((ub >> 16) & 1u);
    return (ua >> 16) | (ub & 0xFFFF0000u);
}

// LDS byte offset of a shared-memory pointer (flat shared aperture: low 32 bits
// of the generic pointer are the LDS address per CDNA5 aperture rules)
static __device__ __forceinline__ uint32_t lds_off_u32(const void* p) {
    return (uint32_t)(uintptr_t)p;
}

// Async DMA: copy 32B (2 x b128) global -> LDS for this lane. Tracked by ASYNCcnt.
static __device__ __forceinline__ void async_copy32(uint32_t lds_byte, const uint32_t* gsrc) {
    asm volatile("global_load_async_to_lds_b128 %0, %1, off\n\t"
                 "global_load_async_to_lds_b128 %0, %1, off offset:16"
                 :: "v"(lds_byte), "v"(gsrc) : "memory");
}
static __device__ __forceinline__ void wait_async0() {
    asm volatile("s_wait_asynccnt 0" ::: "memory");
}

// ===========================================================================
// Swizzle prepasses: build WMMA-lane-layout bf16 operand blocks.
// A-block (16 rows x 32 K) = 1KB: dword t = lane*8+j, lane holds row=lane&15,
//   half=lane>>4, kpair kp = half*4 + (j&3) + 8*(j>=4).
// B-block (16 cols x 32 K) = 1KB: kp = half*4 + (j&3) + 8*(j>=4), lane holds col=lane&15.
// ===========================================================================
__global__ void swizzle_A(const float* __restrict__ src, int lda, int M_real,
                          int MB, int KB, uint32_t* __restrict__ dst)
{
    size_t idx = (size_t)blockIdx.x * blockDim.x + threadIdx.x;
    size_t total = (size_t)MB * KB * 256;
    if (idx >= total) return;
    int t = (int)(idx & 255);
    size_t blk = idx >> 8;
    int kb = (int)(blk % KB);
    int rb = (int)(blk / KB);
    int lane = t >> 3, j = t & 7;
    int row  = rb * 16 + (lane & 15);
    int half = lane >> 4;
    int kp   = half * 4 + (j & 3) + ((j >> 2) << 3);
    int k    = kb * 32 + kp * 2;
    uint32_t v = 0u;
    if (row < M_real) {
        const float* p = src + (size_t)row * lda + k;
        v = pk_bf16(p[0], p[1]);
    }
    dst[idx] = v;
}

__global__ void swizzle_B(const float* __restrict__ src, int ldb, int Ncols,
                          int CB, int KB, uint32_t* __restrict__ dst)
{
    size_t idx = (size_t)blockIdx.x * blockDim.x + threadIdx.x;
    size_t total = (size_t)CB * KB * 256;
    if (idx >= total) return;
    int t = (int)(idx & 255);
    size_t blk = idx >> 8;
    int kb = (int)(blk % KB);
    int cb = (int)(blk / KB);
    int lane = t >> 3, j = t & 7;
    int col  = cb * 16 + (lane & 15);
    int half = lane >> 4;
    int kp   = half * 4 + (j & 3) + ((j >> 2) << 3);
    int k    = kb * 32 + kp * 2;
    uint32_t v = 0u;
    if (col < Ncols)
        v = pk_bf16(src[(size_t)k * ldb + col], src[(size_t)(k + 1) * ldb + col]);
    dst[idx] = v;
}

// ===========================================================================
// gemm_swz: A & B pre-swizzled bf16. Main loop = async DMA copies + WMMA only.
// Block 128x128, 8 wave32s (4M x 2N), wave tile 32x64, double-buffered LDS.
// ===========================================================================
__global__ __launch_bounds__(256)
void gemm_swz(const uint32_t* __restrict__ Asw, const uint32_t* __restrict__ Bsw,
              int KB, const float* __restrict__ bias,
              float* __restrict__ C, int ldc, int Ncols, int M, int do_relu)
{
    __shared__ uint32_t lds[4][2048];   // [0]=A buf0 [1]=B buf0 [2]=A buf1 [3]=B buf1

    const int tid  = threadIdx.x;
    const int lane = tid & 31;
    const int wid  = tid >> 5;
    const int wm   = wid & 3;
    const int wn   = wid >> 2;
    const int half = lane >> 4;
    const int l16  = lane & 15;

    const size_t mb0 = (size_t)blockIdx.y * 8;
    const size_t cb0 = (size_t)blockIdx.x * 8;

    const uint32_t* aSrc = Asw + (mb0 + wid) * (size_t)KB * 256 + lane * 8;
    const uint32_t* bSrc = Bsw + (cb0 + wid) * (size_t)KB * 256 + lane * 8;
    const uint32_t dOff  = (uint32_t)(wid * 256 + lane * 8) * 4u;
    const uint32_t aDst[2] = { lds_off_u32(&lds[0][0]) + dOff, lds_off_u32(&lds[2][0]) + dOff };
    const uint32_t bDst[2] = { lds_off_u32(&lds[1][0]) + dOff, lds_off_u32(&lds[3][0]) + dOff };

    v8f acc[2][4];
    {
        v8f z = {};
        #pragma unroll
        for (int i = 0; i < 2; ++i)
            #pragma unroll
            for (int j = 0; j < 4; ++j) acc[i][j] = z;
    }

    // prologue: DMA tile 0 into buffer 0
    async_copy32(aDst[0], aSrc);
    async_copy32(bDst[0], bSrc);
    wait_async0();
    __syncthreads();

    for (int kb = 0; kb < KB; ++kb) {
        const int cur = kb & 1, nxt = cur ^ 1;
        if (kb + 1 < KB) {
            async_copy32(aDst[nxt], aSrc + (size_t)(kb + 1) * 256);
            async_copy32(bDst[nxt], bSrc + (size_t)(kb + 1) * 256);
        }
        const uint32_t* Al = &lds[cur * 2][0];
        const uint32_t* Bl = &lds[cur * 2 + 1][0];
        v16bf afrag[2], bfrag[4];
        #pragma unroll
        for (int i = 0; i < 2; ++i)
            afrag[i] = *(const v16bf*)(Al + (wm * 2 + i) * 256 + lane * 8);
        #pragma unroll
        for (int j = 0; j < 4; ++j)
            bfrag[j] = *(const v16bf*)(Bl + (wn * 4 + j) * 256 + lane * 8);
        #pragma unroll
        for (int i = 0; i < 2; ++i)
            #pragma unroll
            for (int j = 0; j < 4; ++j)
                acc[i][j] = __builtin_amdgcn_wmma_f32_16x16x32_bf16(
                    false, afrag[i], false, bfrag[j], (short)0, acc[i][j], false, false);
        if (kb + 1 < KB) wait_async0();
        __syncthreads();
    }

    const int m0 = blockIdx.y * TILE_M;
    const int n0 = blockIdx.x * TILE_N;
    #pragma unroll
    for (int j = 0; j < 4; ++j) {
        const int n = n0 + wn * 64 + j * 16 + l16;
        const float bv = (n < Ncols) ? bias[n] : 0.0f;
        #pragma unroll
        for (int i = 0; i < 2; ++i)
            #pragma unroll
            for (int v = 0; v < 8; ++v) {
                const int r = m0 + wm * 32 + i * 16 + v + 8 * half;
                float val = acc[i][j][v] + bv;
                if (do_relu) val = fmaxf(val, 0.0f);
                if (r < M && n < Ncols) C[(size_t)r * ldc + n] = val;
            }
    }
}

// ===========================================================================
// gemm_convA: A fp32 (converted in-loop, co-executes with XDL), B pre-swizzled
// bf16 via async DMA. Used for GEMM1 (x @ W6). Writes relu(C) to strided out.
// ===========================================================================
__global__ __launch_bounds__(256)
void gemm_convA(const float* __restrict__ A, int lda, int M,
                const uint32_t* __restrict__ Bsw, int KB,
                const float* __restrict__ bias,
                float* __restrict__ C, int ldc, int Ncols, int do_relu)
{
    __shared__ uint32_t lds[4][2048];

    const int tid  = threadIdx.x;
    const int lane = tid & 31;
    const int wid  = tid >> 5;
    const int wm   = wid & 3;
    const int wn   = wid >> 2;
    const int half = lane >> 4;
    const int l16  = lane & 15;

    const int m0 = blockIdx.y * TILE_M;
    const size_t cb0 = (size_t)blockIdx.x * 8;

    // A staging: thread -> (row, 16-K half)
    const int a_row = tid >> 1;
    const int a_kh  = (tid & 1) * 16;
    const int rg    = m0 + a_row;
    const bool aValid = rg < M;
    const float* aPtr = A + (size_t)rg * lda + a_kh;

    const uint32_t* bSrc = Bsw + (cb0 + wid) * (size_t)KB * 256 + lane * 8;
    const uint32_t dOff  = (uint32_t)(wid * 256 + lane * 8) * 4u;
    const uint32_t bDst[2] = { lds_off_u32(&lds[1][0]) + dOff, lds_off_u32(&lds[3][0]) + dOff };

    // A LDS store targets (swizzled layout) for both buffers
    uint32_t* aSt0[2]; uint32_t* aSt1[2];
    #pragma unroll
    for (int b = 0; b < 2; ++b) {
        uint32_t* base = &lds[b * 2][(a_row >> 4) * 256];
        aSt0[b] = base + (a_row & 15) * 8 + (a_kh >> 2);          // half'=0 dwords
        aSt1[b] = base + ((a_row & 15) + 16) * 8 + (a_kh >> 2);   // half'=1 dwords
    }

    v8f acc[2][4];
    {
        v8f z = {};
        #pragma unroll
        for (int i = 0; i < 2; ++i)
            #pragma unroll
            for (int j = 0; j < 4; ++j) acc[i][j] = z;
    }

    // prologue: tile 0
    {
        uint32_t buf[8];
        if (aValid) {
            const float4* p = (const float4*)aPtr;
            float4 f0 = p[0], f1 = p[1], f2 = p[2], f3 = p[3];
            buf[0] = pk_bf16(f0.x, f0.y); buf[1] = pk_bf16(f0.z, f0.w);
            buf[2] = pk_bf16(f1.x, f1.y); buf[3] = pk_bf16(f1.z, f1.w);
            buf[4] = pk_bf16(f2.x, f2.y); buf[5] = pk_bf16(f2.z, f2.w);
            buf[6] = pk_bf16(f3.x, f3.y); buf[7] = pk_bf16(f3.z, f3.w);
        } else {
            #pragma unroll
            for (int q = 0; q < 8; ++q) buf[q] = 0u;
        }
        async_copy32(bDst[0], bSrc);
        ((uint4*)aSt0[0])[0] = make_uint4(buf[0], buf[1], buf[2], buf[3]);
        ((uint4*)aSt1[0])[0] = make_uint4(buf[4], buf[5], buf[6], buf[7]);
        wait_async0();
    }
    __syncthreads();

    for (int kb = 0; kb < KB; ++kb) {
        const int cur = kb & 1, nxt = cur ^ 1;
        float4 f0, f1, f2, f3;
        const bool more = (kb + 1 < KB);
        if (more) {
            if (aValid) {
                const float4* p = (const float4*)(aPtr + (size_t)(kb + 1) * TILE_K);
                f0 = p[0]; f1 = p[1]; f2 = p[2]; f3 = p[3];
            }
            async_copy32(bDst[nxt], bSrc + (size_t)(kb + 1) * 256);
        }

        const uint32_t* Al = &lds[cur * 2][0];
        const uint32_t* Bl = &lds[cur * 2 + 1][0];
        v16bf afrag[2], bfrag[4];
        #pragma unroll
        for (int i = 0; i < 2; ++i)
            afrag[i] = *(const v16bf*)(Al + (wm * 2 + i) * 256 + lane * 8);
        #pragma unroll
        for (int j = 0; j < 4; ++j)
            bfrag[j] = *(const v16bf*)(Bl + (wn * 4 + j) * 256 + lane * 8);
        #pragma unroll
        for (int i = 0; i < 2; ++i)
            #pragma unroll
            for (int j = 0; j < 4; ++j)
                acc[i][j] = __builtin_amdgcn_wmma_f32_16x16x32_bf16(
                    false, afrag[i], false, bfrag[j], (short)0, acc[i][j], false, false);

        if (more) {
            uint32_t buf[8];
            if (aValid) {
                buf[0] = pk_bf16(f0.x, f0.y); buf[1] = pk_bf16(f0.z, f0.w);
                buf[2] = pk_bf16(f1.x, f1.y); buf[3] = pk_bf16(f1.z, f1.w);
                buf[4] = pk_bf16(f2.x, f2.y); buf[5] = pk_bf16(f2.z, f2.w);
                buf[6] = pk_bf16(f3.x, f3.y); buf[7] = pk_bf16(f3.z, f3.w);
            } else {
                #pragma unroll
                for (int q = 0; q < 8; ++q) buf[q] = 0u;
            }
            ((uint4*)aSt0[nxt])[0] = make_uint4(buf[0], buf[1], buf[2], buf[3]);
            ((uint4*)aSt1[nxt])[0] = make_uint4(buf[4], buf[5], buf[6], buf[7]);
            wait_async0();
        }
        __syncthreads();
    }

    const int n0 = blockIdx.x * TILE_N;
    #pragma unroll
    for (int j = 0; j < 4; ++j) {
        const int n = n0 + wn * 64 + j * 16 + l16;
        const float bv = (n < Ncols) ? bias[n] : 0.0f;
        #pragma unroll
        for (int i = 0; i < 2; ++i)
            #pragma unroll
            for (int v = 0; v < 8; ++v) {
                const int r = m0 + wm * 32 + i * 16 + v + 8 * half;
                float val = acc[i][j][v] + bv;
                if (do_relu) val = fmaxf(val, 0.0f);
                if (r < M && n < Ncols) C[(size_t)r * ldc + n] = val;
            }
    }
}

// ===========================================================================
// Legacy fallback GEMM (everything fused, small workspace) — from round 1.
// ===========================================================================
__global__ __launch_bounds__(256)
void gemm_bias_act(const float* __restrict__ A, int lda,
                   const float* __restrict__ B, int ldb, int Ncols,
                   const float* __restrict__ bias,
                   float* __restrict__ C, int ldc,
                   float* __restrict__ C2, int ldc2,
                   int M, int K, int do_relu)
{
    __shared__ uint32_t As[TILE_M][LDS_STRIDE];
    __shared__ uint32_t Bs[TILE_N][LDS_STRIDE];

    const int tid  = threadIdx.x;
    const int lane = tid & 31;
    const int wid  = tid >> 5;
    const int wm   = wid & 3;
    const int wn   = wid >> 2;
    const int half = lane >> 4;
    const int l16  = lane & 15;
    const int m0 = blockIdx.y * TILE_M;
    const int n0 = blockIdx.x * TILE_N;
    const int a_row = tid >> 1;
    const int a_kh  = (tid & 1) * 16;
    const int b_c4  = (tid & 31) * 4;
    const int b_k   = (tid >> 5) * 4;

    v8f acc[2][4];
    {
        v8f z = {};
        #pragma unroll
        for (int i = 0; i < 2; ++i)
            #pragma unroll
            for (int j = 0; j < 4; ++j) acc[i][j] = z;
    }

    for (int k0 = 0; k0 < K; k0 += TILE_K) {
        {
            uint32_t buf[8];
            const int rg2 = m0 + a_row;
            if (rg2 < M) {
                const float4* ap = (const float4*)(A + (size_t)rg2 * lda + k0 + a_kh);
                float4 f0 = ap[0], f1 = ap[1], f2 = ap[2], f3 = ap[3];
                buf[0] = pk_bf16(f0.x, f0.y); buf[1] = pk_bf16(f0.z, f0.w);
                buf[2] = pk_bf16(f1.x, f1.y); buf[3] = pk_bf16(f1.z, f1.w);
                buf[4] = pk_bf16(f2.x, f2.y); buf[5] = pk_bf16(f2.z, f2.w);
                buf[6] = pk_bf16(f3.x, f3.y); buf[7] = pk_bf16(f3.z, f3.w);
            } else {
                #pragma unroll
                for (int q = 0; q < 8; ++q) buf[q] = 0u;
            }
            uint32_t* dst = &As[a_row][a_kh >> 1];
            ((uint4*)dst)[0] = make_uint4(buf[0], buf[1], buf[2], buf[3]);
            ((uint4*)dst)[1] = make_uint4(buf[4], buf[5], buf[6], buf[7]);
        }
        {
            float4 r0, r1, r2, r3;
            const int cg = n0 + b_c4;
            if (cg + 3 < Ncols) {
                r0 = *(const float4*)(B + (size_t)(k0 + b_k + 0) * ldb + cg);
                r1 = *(const float4*)(B + (size_t)(k0 + b_k + 1) * ldb + cg);
                r2 = *(const float4*)(B + (size_t)(k0 + b_k + 2) * ldb + cg);
                r3 = *(const float4*)(B + (size_t)(k0 + b_k + 3) * ldb + cg);
            } else {
                float t[4][4];
                #pragma unroll
                for (int kk = 0; kk < 4; ++kk)
                    #pragma unroll
                    for (int jj = 0; jj < 4; ++jj)
                        t[kk][jj] = (cg + jj < Ncols)
                                  ? B[(size_t)(k0 + b_k + kk) * ldb + cg + jj] : 0.0f;
                r0 = make_float4(t[0][0], t[0][1], t[0][2], t[0][3]);
                r1 = make_float4(t[1][0], t[1][1], t[1][2], t[1][3]);
                r2 = make_float4(t[2][0], t[2][1], t[2][2], t[2][3]);
                r3 = make_float4(t[3][0], t[3][1], t[3][2], t[3][3]);
            }
            const int kp = b_k >> 1;
            Bs[b_c4 + 0][kp + 0] = pk_bf16(r0.x, r1.x);
            Bs[b_c4 + 0][kp + 1] = pk_bf16(r2.x, r3.x);
            Bs[b_c4 + 1][kp + 0] = pk_bf16(r0.y, r1.y);
            Bs[b_c4 + 1][kp + 1] = pk_bf16(r2.y, r3.y);
            Bs[b_c4 + 2][kp + 0] = pk_bf16(r0.z, r1.z);
            Bs[b_c4 + 2][kp + 1] = pk_bf16(r2.z, r3.z);
            Bs[b_c4 + 3][kp + 0] = pk_bf16(r0.w, r1.w);
            Bs[b_c4 + 3][kp + 1] = pk_bf16(r2.w, r3.w);
        }
        __syncthreads();

        v16bf afrag[2], bfrag[4];
        #pragma unroll
        for (int i = 0; i < 2; ++i) {
            const uint32_t* row = &As[wm * 32 + i * 16 + l16][0];
            uint32_t* ap = (uint32_t*)&afrag[i];
            ((uint4*)ap)[0] = *(const uint4*)&row[half * 4];
            ((uint4*)ap)[1] = *(const uint4*)&row[half * 4 + 8];
        }
        #pragma unroll
        for (int j = 0; j < 4; ++j) {
            const uint32_t* row = &Bs[wn * 64 + j * 16 + l16][0];
            uint32_t* bp = (uint32_t*)&bfrag[j];
            ((uint4*)bp)[0] = *(const uint4*)&row[half * 4];
            ((uint4*)bp)[1] = *(const uint4*)&row[half * 4 + 8];
        }
        #pragma unroll
        for (int i = 0; i < 2; ++i)
            #pragma unroll
            for (int j = 0; j < 4; ++j)
                acc[i][j] = __builtin_amdgcn_wmma_f32_16x16x32_bf16(
                    false, afrag[i], false, bfrag[j], (short)0, acc[i][j], false, false);
        __syncthreads();
    }

    #pragma unroll
    for (int j = 0; j < 4; ++j) {
        const int n = n0 + wn * 64 + j * 16 + l16;
        const float bv = (n < Ncols) ? bias[n] : 0.0f;
        #pragma unroll
        for (int i = 0; i < 2; ++i)
            #pragma unroll
            for (int v = 0; v < 8; ++v) {
                const int r = m0 + wm * 32 + i * 16 + v + 8 * half;
                float val = acc[i][j][v] + bv;
                if (do_relu) val = fmaxf(val, 0.0f);
                if (r < M && n < Ncols) {
                    C[(size_t)r * ldc + n] = val;
                    if (C2) C2[(size_t)r * ldc2 + n] = val;
                }
            }
    }
}

// ===========================================================================
// Shared small kernels
// ===========================================================================
__global__ void pack_heads(const float* __restrict__ Wcls, const float* __restrict__ Wbox,
                           const float* __restrict__ bcls, const float* __restrict__ bboxb,
                           float* __restrict__ Whead, float* __restrict__ bhead)
{
    const int NH = 416;
    const int idx = blockIdx.x * blockDim.x + threadIdx.x;
    if (idx < 1024 * NH) {
        const int k = idx / NH, j = idx % NH;
        float v = 0.0f;
        if (j < 81)       v = Wcls[k * 81 + j];
        else if (j < 405) v = Wbox[k * 324 + (j - 81)];
        Whead[idx] = v;
    }
    if (idx < NH) {
        float v = 0.0f;
        if (idx < 81)       v = bcls[idx];
        else if (idx < 405) v = bboxb[idx - 81];
        bhead[idx] = v;
    }
}

__global__ __launch_bounds__(256)
void postprocess(const float* __restrict__ heads, const float* __restrict__ boxes,
                 float* __restrict__ out, int R)
{
    const int lane = threadIdx.x & 31;
    const int row  = blockIdx.x * 8 + (threadIdx.x >> 5);
    if (row >= R) return;
    const float* hr = heads + (size_t)row * 416;

    const float b0 = boxes[row * 4 + 0], b1 = boxes[row * 4 + 1];
    const float b2 = boxes[row * 4 + 2], b3 = boxes[row * 4 + 3];
    const float w  = b2 - b0 + 1.0f, h = b3 - b1 + 1.0f;
    const float cx = b0 + 0.5f * w, cy = b1 + 0.5f * h;

    float lg[3];
    float mx = -3.4e38f;
    #pragma unroll
    for (int t = 0; t < 3; ++t) {
        const int c = lane + t * 32;
        lg[t] = (c < 81) ? hr[c] : -3.4e38f;
        mx = fmaxf(mx, lg[t]);
    }
    #pragma unroll
    for (int off = 16; off > 0; off >>= 1) mx = fmaxf(mx, __shfl_xor(mx, off, 32));
    float e[3], s = 0.0f;
    #pragma unroll
    for (int t = 0; t < 3; ++t) {
        const int c = lane + t * 32;
        e[t] = (c < 81) ? expf(lg[t] - mx) : 0.0f;
        s += e[t];
    }
    #pragma unroll
    for (int off = 16; off > 0; off >>= 1) s += __shfl_xor(s, off, 32);
    const float inv = 1.0f / s;

    const float CLIP = 4.135166556742356f;
    float* orow = out + (size_t)row * 1424;
    #pragma unroll
    for (int t = 0; t < 3; ++t) {
        const int c = lane + t * 32;
        if (c >= 1 && c < 81) {
            const float p = e[t] * inv;
            const bool keep = p > 0.05f;
            orow[c - 1] = keep ? p : 0.0f;
            const float dx = hr[81 + c * 4 + 0] * 0.1f;
            const float dy = hr[81 + c * 4 + 1] * 0.1f;
            const float dw = fminf(hr[81 + c * 4 + 2] * 0.2f, CLIP);
            const float dh = fminf(hr[81 + c * 4 + 3] * 0.2f, CLIP);
            const float pcx = dx * w + cx, pcy = dy * h + cy;
            const float pw  = expf(dw) * w, ph = expf(dh) * h;
            const float x0 = pcx - 0.5f * pw, y0 = pcy - 0.5f * ph;
            const float x1 = pcx + 0.5f * pw - 1.0f, y1 = pcy + 0.5f * ph - 1.0f;
            const int bo = 80 + (c - 1) * 4;
            orow[bo + 0] = keep ? x0 : 0.0f;
            orow[bo + 1] = keep ? y0 : 0.0f;
            orow[bo + 2] = keep ? x1 : 0.0f;
            orow[bo + 3] = keep ? y1 : 0.0f;
        }
    }
}

extern "C" void kernel_launch(void* const* d_in, const int* in_sizes, int n_in,
                              void* d_out, int out_size, void* d_ws, size_t ws_size,
                              hipStream_t stream)
{
    (void)in_sizes; (void)n_in; (void)out_size;
    const float* x     = (const float*)d_in[0];
    const float* boxes = (const float*)d_in[1];
    const float* W6    = (const float*)d_in[2];
    const float* b6    = (const float*)d_in[3];
    const float* W7    = (const float*)d_in[4];
    const float* b7    = (const float*)d_in[5];
    const float* Wcls  = (const float*)d_in[6];
    const float* bcls  = (const float*)d_in[7];
    const float* Wbox  = (const float*)d_in[8];
    const float* bboxb = (const float*)d_in[9];
    float* out = (float*)d_out;

    const int R = 2000, D = 12544, H = 1024, NH = 416;
    const int KB1 = D / 32;   // 392
    const int KB2 = H / 32;   // 32

    // fast-path workspace layout
    const size_t szW6sw = (size_t)(H / 16) * KB1 * 1024;   // 25.7 MB
    const size_t szW7sw = (size_t)(H / 16) * KB2 * 1024;   // 2.1 MB
    const size_t szWHsw = (size_t)32 * KB2 * 1024;         // 1.05 MB (512 padded cols)
    const size_t szF6sw = (size_t)128 * KB2 * 1024;        // 4.2 MB (2048 padded rows)
    const size_t szF7sw = szF6sw;
    const size_t szWh   = (size_t)H * NH * 4;
    const size_t szBh   = (size_t)NH * 4;
    const size_t szF7   = (size_t)R * H * 4;
    const size_t szHd   = (size_t)R * NH * 4;
    const size_t FASTREQ = szW6sw + szW7sw + szWHsw + szF6sw + szF7sw + szWh + szBh + szF7 + szHd;

    if (ws_size >= FASTREQ) {
        char* p = (char*)d_ws;
        uint32_t* W6sw = (uint32_t*)p; p += szW6sw;
        uint32_t* W7sw = (uint32_t*)p; p += szW7sw;
        uint32_t* WHsw = (uint32_t*)p; p += szWHsw;
        uint32_t* F6sw = (uint32_t*)p; p += szF6sw;
        uint32_t* F7sw = (uint32_t*)p; p += szF7sw;
        float* Whead = (float*)p; p += szWh;
        float* bhead = (float*)p; p += szBh;
        float* f7    = (float*)p; p += szF7;
        float* heads = (float*)p; p += szHd;

        pack_heads<<<(H * NH + 255) / 256, 256, 0, stream>>>(Wcls, Wbox, bcls, bboxb, Whead, bhead);

        // one-time weight swizzles (bf16, WMMA lane layout)
        {
            size_t n = (size_t)(H / 16) * KB1 * 256;
            swizzle_B<<<(unsigned)((n + 255) / 256), 256, 0, stream>>>(W6, H, H, H / 16, KB1, W6sw);
        }
        {
            size_t n = (size_t)(H / 16) * KB2 * 256;
            swizzle_B<<<(unsigned)((n + 255) / 256), 256, 0, stream>>>(W7, H, H, H / 16, KB2, W7sw);
        }
        {
            size_t n = (size_t)32 * KB2 * 256;
            swizzle_B<<<(unsigned)((n + 255) / 256), 256, 0, stream>>>(Whead, NH, NH, 32, KB2, WHsw);
        }

        // GEMM1: relu(x @ W6 + b6) -> out[:, 400:1424] (strided)
        dim3 g1(H / TILE_N, (R + TILE_M - 1) / TILE_M);                 // 8 x 16
        gemm_convA<<<g1, 256, 0, stream>>>(x, D, R, W6sw, KB1, b6, out + 400, 1424, H, 1);

        // swizzle f6 (read back from out, strided) -> F6sw
        {
            size_t n = (size_t)128 * KB2 * 256;
            swizzle_A<<<(unsigned)((n + 255) / 256), 256, 0, stream>>>(out + 400, 1424, R, 128, KB2, F6sw);
        }
        // GEMM2: relu(f6 @ W7 + b7) -> f7
        gemm_swz<<<g1, 256, 0, stream>>>(F6sw, W7sw, KB2, b7, f7, H, H, R, 1);
        // swizzle f7 -> F7sw
        {
            size_t n = (size_t)128 * KB2 * 256;
            swizzle_A<<<(unsigned)((n + 255) / 256), 256, 0, stream>>>(f7, H, R, 128, KB2, F7sw);
        }
        // GEMM3: f7 @ Whead + bhead -> heads
        dim3 g3(4, (R + TILE_M - 1) / TILE_M);                          // 512 padded cols
        gemm_swz<<<g3, 256, 0, stream>>>(F7sw, WHsw, KB2, bhead, heads, NH, NH, R, 0);

        postprocess<<<(R + 7) / 8, 256, 0, stream>>>(heads, boxes, out, R);
    } else {
        // legacy small-workspace fallback
        float* ws    = (float*)d_ws;
        float* f6    = ws;
        float* f7    = f6 + (size_t)R * H;
        float* heads = f7 + (size_t)R * H;
        float* Whead = heads + (size_t)R * NH;
        float* bhead = Whead + (size_t)H * NH;

        pack_heads<<<(H * NH + 255) / 256, 256, 0, stream>>>(Wcls, Wbox, bcls, bboxb, Whead, bhead);
        dim3 g1(H / TILE_N, (R + TILE_M - 1) / TILE_M);
        gemm_bias_act<<<g1, 256, 0, stream>>>(x, D, W6, H, H, b6, f6, H, out + 400, 1424, R, D, 1);
        gemm_bias_act<<<g1, 256, 0, stream>>>(f6, H, W7, H, H, b7, f7, H, nullptr, 0, R, H, 1);
        dim3 g3((NH + TILE_N - 1) / TILE_N, (R + TILE_M - 1) / TILE_M);
        gemm_bias_act<<<g3, 256, 0, stream>>>(f7, H, Whead, NH, NH, bhead, heads, NH, nullptr, 0, R, H, 0);
        postprocess<<<(R + 7) / 8, 256, 0, stream>>>(heads, boxes, out, R);
    }
}
